// MLP_87952340288005
// MI455X (gfx1250) — matrix operand order repaired and checked by reading, not compiled, not run on hardware
//
#include <hip/hip_runtime.h>
#include <stdint.h>

// ---------------------------------------------------------------------------
// Problem constants (from the reference)
// ---------------------------------------------------------------------------
constexpr int BATCH = 16;
constexpr int NOBJ  = 64;
constexpr int PPAIR = 2016;
constexpr int EFDIM = 1024;   // interaction feature dim == node feature dim
constexpr int MDIM  = 512;    // edge/node linear output dim
constexpr int KDIM  = 1024;   // CI dim (2*MDIM) == EFDIM
constexpr int HID   = 512;
constexpr int ROWS  = BATCH * PPAIR;        // 32256 (divisible by 64)
constexpr int OUTC  = 17;                   // 9 + 5 + 3

// WMMA vector types (wave32, gfx1250)
typedef __attribute__((ext_vector_type(16))) __bf16 v16bf;
typedef __attribute__((ext_vector_type(8)))  float  v8f;

union Frag {
    uint4 q[2];
    v16bf v;
};

__device__ __forceinline__ uint16_t f2bf(float f) {
    uint32_t u = __float_as_uint(f);
    uint32_t r = u + 0x7FFFu + ((u >> 16) & 1u);   // round-to-nearest-even
    return (uint16_t)(r >> 16);
}
__device__ __forceinline__ float bf2f(uint16_t h) {
    return __uint_as_float(((uint32_t)h) << 16);
}
__device__ __forceinline__ uint32_t pk2(float a, float b) {
    return (uint32_t)f2bf(a) | ((uint32_t)f2bf(b) << 16);
}

// A fragment (16x32 bf16): two b128 loads, chunks at +0 and +16 halves
__device__ __forceinline__ void ldA(Frag& f, const uint16_t* p) {
    const uint4* q = (const uint4*)p;
    f.q[0] = q[0];
    f.q[1] = q[2];
}
// B fragment from the prepacked stream: 32 contiguous bytes per lane
__device__ __forceinline__ void ldB(Frag& f, const uint16_t* p) {
    const uint4* q = (const uint4*)p;
    f.q[0] = q[0];
    f.q[1] = q[1];
}

// ---------------------------------------------------------------------------
// Pack 5 weights [K=1024, N=512] fp32 -> bf16 WMMA-B fragment order:
//   dst[((w*32 + nt)*32 + kt)*32 + lane][16 halves]
// ---------------------------------------------------------------------------
__global__ void pack_w(const float* __restrict__ w0, const float* __restrict__ w1,
                       const float* __restrict__ w2, const float* __restrict__ w3,
                       const float* __restrict__ w4, uint16_t* __restrict__ dst) {
    int tid  = blockIdx.x * 256 + threadIdx.x;       // 5*32*32*32 = 163840
    int lane = tid & 31;
    int kt   = (tid >> 5) & 31;
    int nt   = (tid >> 10) & 31;
    int w    = tid >> 15;
    if (w >= 5) return;
    const float* W = (w == 0) ? w0 : (w == 1) ? w1 : (w == 2) ? w2 : (w == 3) ? w3 : w4;
    int ncol  = nt * 16 + (lane & 15);
    int kbase = kt * 32 + (lane >> 4) * 8;
    size_t o = ((((size_t)w * 32 + nt) * 32 + kt) * 32 + lane) * 16;
#pragma unroll
    for (int h = 0; h < 16; ++h)
        dst[o + h] = f2bf(W[(size_t)(kbase + h + (h >= 8 ? 8 : 0)) * MDIM + ncol]);
}

// concat the three layer-1 biases -> [1536] fp32
__global__ void b1cat_k(const float* __restrict__ a, const float* __restrict__ b,
                        const float* __restrict__ c, float* __restrict__ dst) {
    int i = blockIdx.x * 256 + threadIdx.x;
    if (i >= 3 * HID) return;
    dst[i] = (i < 512) ? a[i] : (i < 1024) ? b[i - 512] : c[i - 1024];
}

// ---------------------------------------------------------------------------
// Gather: xg[g,:] = 0.5*(IF[b,i0,i1,:] + IF[b,i1,i0,:])  -> bf16
//         ng[g,:] = 0.5*(CNF[b,i0,:]   + CNF[b,i1,:])    -> bf16
// ---------------------------------------------------------------------------
__global__ void gather_k(const float* __restrict__ IF, const float* __restrict__ CNF,
                         const int* __restrict__ pairs,
                         uint16_t* __restrict__ xg, uint16_t* __restrict__ ng) {
    int g = blockIdx.x;                 // 0 .. 32255
    int b = g / PPAIR;
    int i0 = pairs[(size_t)g * 2 + 0];
    int i1 = pairs[(size_t)g * 2 + 1];
    const float4* e0 = (const float4*)(IF + (((size_t)b * NOBJ + i0) * NOBJ + i1) * EFDIM);
    const float4* e1 = (const float4*)(IF + (((size_t)b * NOBJ + i1) * NOBJ + i0) * EFDIM);
    const float4* n0 = (const float4*)(CNF + ((size_t)b * NOBJ + i0) * EFDIM);
    const float4* n1 = (const float4*)(CNF + ((size_t)b * NOBJ + i1) * EFDIM);
    int t = threadIdx.x;                // 256 threads * 4 floats = 1024
    {
        float4 a = e0[t], c = e1[t];
        uint2 st = { pk2(0.5f * (a.x + c.x), 0.5f * (a.y + c.y)),
                     pk2(0.5f * (a.z + c.z), 0.5f * (a.w + c.w)) };
        ((uint2*)(xg + (size_t)g * KDIM))[t] = st;
    }
    {
        float4 a = n0[t], c = n1[t];
        uint2 st = { pk2(0.5f * (a.x + c.x), 0.5f * (a.y + c.y)),
                     pk2(0.5f * (a.z + c.z), 0.5f * (a.w + c.w)) };
        ((uint2*)(ng + (size_t)g * KDIM))[t] = st;
    }
}

// ---------------------------------------------------------------------------
// WMMA GEMM, register-double-buffered.
// Block = 256 threads = 8 waves arranged 2(Mrow) x 4(Ncol).
// Wave tile: 32M x 64N (8 v8f accumulators). Workgroup tile: 64M x 256N.
// grid = (ROWS/64, N/256). K fixed at 1024 (32 steps of 32).
// Loads for step kt+1 are issued before the WMMAs of step kt. WMMAs are
// ordered j-outer/mi-inner so BOTH A fragments stay live across the quad
// sequence (prevents the allocator from folding them into one register set
// and re-loading mid-iteration behind an s_wait_loadcnt 0).
// ---------------------------------------------------------------------------
__global__ void __launch_bounds__(256)
gemm_wmma(const uint16_t* __restrict__ A,
          const uint16_t* __restrict__ Bpack,
          const float* __restrict__ bias,
          const int* __restrict__ nrel,
          uint16_t* __restrict__ C,
          int ldc, int colOff, int doRelu) {
    const int wid  = threadIdx.x >> 5;
    const int lane = threadIdx.x & 31;
    const int wrow = wid >> 2;                       // 0..1
    const int wcol = wid & 3;                        // 0..3
    const int mBase   = blockIdx.x * 64 + wrow * 32;
    const int ntBase  = blockIdx.y * 16 + wcol * 4;  // fragment-tile index in Bpack
    const int colBase = colOff + ntBase * 16;
    const int chunk   = lane >> 4;

    // Per-lane base pointers (A: lane&15 = M within 16-row tile; chunk = K half)
    const uint16_t* ap0 = A + (size_t)(mBase + (lane & 15)) * KDIM + chunk * 8;
    const uint16_t* ap1 = ap0 + (size_t)16 * KDIM;
    // B: [nt][kt][lane][16 halves]; per-(j,kt) offset = (j*32+kt)*512 halves
    const uint16_t* bb  = Bpack + ((size_t)ntBase * 32 * 32 + lane) * 16;

    v8f acc[2][4];
#pragma unroll
    for (int mi = 0; mi < 2; ++mi)
#pragma unroll
        for (int j = 0; j < 4; ++j)
            acc[mi][j] = (v8f){0.f,0.f,0.f,0.f,0.f,0.f,0.f,0.f};

    Frag aF[2][2];   // [buf][mi]
    Frag bF[2][4];   // [buf][j]

#define LOAD_SET(buf, kt)                                                     \
    {                                                                         \
        ldA(aF[buf][0], ap0 + (kt) * 32);                                     \
        ldA(aF[buf][1], ap1 + (kt) * 32);                                     \
        _Pragma("unroll")                                                     \
        for (int j = 0; j < 4; ++j)                                           \
            ldB(bF[buf][j], bb + ((size_t)(j * 32 + (kt))) * 512);            \
    }

    // j outer, mi inner: keeps aF[buf][0] AND aF[buf][1] live through the
    // whole sequence so their loads cluster with the B loads.
#define COMPUTE(buf)                                                          \
    {                                                                         \
        _Pragma("unroll")                                                     \
        for (int j = 0; j < 4; ++j)                                           \
            _Pragma("unroll")                                                 \
            for (int mi = 0; mi < 2; ++mi)                                    \
                acc[mi][j] = __builtin_amdgcn_wmma_f32_16x16x32_bf16(         \
                    false, aF[buf][mi].v, false, bF[buf][j].v,                \
                    (short)0, acc[mi][j], false, false);                      \
    }

    LOAD_SET(0, 0)
#pragma unroll 1
    for (int kt = 0; kt < 32; kt += 2) {
        LOAD_SET(1, kt + 1)          // prefetch odd step into registers
        // distant prefetch of the A stream (lowers to global_prefetch_b8)
        if (kt + 8 < 32) {
            __builtin_prefetch(ap0 + (kt + 8) * 32, 0, 0);
            __builtin_prefetch(ap1 + (kt + 8) * 32, 0, 0);
        }
        COMPUTE(0)                   // compute even step
        if (kt + 2 < 32) LOAD_SET(0, kt + 2)   // prefetch next even step
        COMPUTE(1)                   // compute odd step
    }
#undef LOAD_SET
#undef COMPUTE

    // C/D layout: VGPR r -> M = (lane>>4)*8 + r, N = lane&15
    const int rhalf = lane >> 4;
    const int ncl   = lane & 15;
#pragma unroll
    for (int mi = 0; mi < 2; ++mi) {
#pragma unroll
        for (int j = 0; j < 4; ++j) {
            const int   col = colBase + j * 16 + ncl;
            const float bv  = bias[ntBase * 16 + j * 16 + ncl];
#pragma unroll
            for (int r = 0; r < 8; ++r) {
                const int row = mBase + mi * 16 + rhalf * 8 + r;
                float v = acc[mi][j][r] + bv;
                if (doRelu) v = fmaxf(v, 0.f);
                if (nrel) {
                    int b = row / PPAIR;
                    int p = row - b * PPAIR;
                    if (p >= nrel[b]) v = 0.f;
                }
                C[(size_t)row * ldc + col] = f2bf(v);
            }
        }
    }
}

// ---------------------------------------------------------------------------
// Final [512 -> 17] layer, one wave per row (lanes 0..16 = output columns).
// H is [ROWS, 1536] bf16 (three 512-wide heads).
// ---------------------------------------------------------------------------
__global__ void head_out(const uint16_t* __restrict__ H,
                         const float* __restrict__ w0, const float* __restrict__ b0,
                         const float* __restrict__ w1, const float* __restrict__ b1,
                         const float* __restrict__ w2, const float* __restrict__ b2,
                         float* __restrict__ out) {
    int row  = blockIdx.x;
    int lane = threadIdx.x;
    if (lane >= OUTC) return;
    int h, lc, nh;
    const float* W; const float* bb;
    if (lane < 9)       { h = 0; lc = lane;      nh = 9; W = w0; bb = b0; }
    else if (lane < 14) { h = 1; lc = lane - 9;  nh = 5; W = w1; bb = b1; }
    else                { h = 2; lc = lane - 14; nh = 3; W = w2; bb = b2; }
    const uint16_t* hr = H + (size_t)row * (3 * HID) + h * HID;
    float acc = 0.f;
#pragma unroll 4
    for (int k4 = 0; k4 < HID / 4; ++k4) {
        uint2 hv = ((const uint2*)hr)[k4];
        int k = k4 * 4;
        acc = fmaf(bf2f((uint16_t)(hv.x & 0xFFFF)),  W[(k + 0) * nh + lc], acc);
        acc = fmaf(bf2f((uint16_t)(hv.x >> 16)),     W[(k + 1) * nh + lc], acc);
        acc = fmaf(bf2f((uint16_t)(hv.y & 0xFFFF)),  W[(k + 2) * nh + lc], acc);
        acc = fmaf(bf2f((uint16_t)(hv.y >> 16)),     W[(k + 3) * nh + lc], acc);
    }
    out[(size_t)row * OUTC + lane] = acc + bb[lc];
}

// ---------------------------------------------------------------------------
// Launch
// ---------------------------------------------------------------------------
extern "C" void kernel_launch(void* const* d_in, const int* in_sizes, int n_in,
                              void* d_out, int out_size, void* d_ws, size_t ws_size,
                              hipStream_t stream) {
    const float* IF    = (const float*)d_in[0];
    const float* CNF   = (const float*)d_in[1];
    const int*   pairs = (const int*)  d_in[2];
    const int*   nrel  = (const int*)  d_in[3];
    const float* edgeW = (const float*)d_in[4];
    const float* edgeB = (const float*)d_in[5];
    const float* nodeW = (const float*)d_in[6];
    const float* nodeB = (const float*)d_in[7];
    const float* lrW1  = (const float*)d_in[8];
    const float* lrB1  = (const float*)d_in[9];
    const float* lrW2  = (const float*)d_in[10];
    const float* lrB2  = (const float*)d_in[11];
    const float* crW1  = (const float*)d_in[12];
    const float* crB1  = (const float*)d_in[13];
    const float* crW2  = (const float*)d_in[14];
    const float* crB2  = (const float*)d_in[15];
    const float* mrW1  = (const float*)d_in[16];
    const float* mrB1  = (const float*)d_in[17];
    const float* mrW2  = (const float*)d_in[18];
    const float* mrB2  = (const float*)d_in[19];
    float* out = (float*)d_out;

    // workspace layout (~289 MB total)
    const size_t REGION_HALVES = (size_t)32 * 32 * 32 * 16;     // per-weight pack: 1 MB
    char* ws = (char*)d_ws;
    size_t off = 0;
    uint16_t* pack = (uint16_t*)(ws + off); off += 5 * REGION_HALVES * 2;
    float*    b1c  = (float*)(ws + off);    off += 3 * HID * sizeof(float);
    off = (off + 255) & ~(size_t)255;
    uint16_t* xg = (uint16_t*)(ws + off);   off += (size_t)ROWS * KDIM * 2;
    uint16_t* ng = (uint16_t*)(ws + off);   off += (size_t)ROWS * KDIM * 2;
    uint16_t* ci = (uint16_t*)(ws + off);   off += (size_t)ROWS * KDIM * 2;
    uint16_t* H  = (uint16_t*)(ws + off);   off += (size_t)ROWS * (3 * HID) * 2;
    (void)ws_size; (void)in_sizes; (void)n_in; (void)out_size;

    // 1) pack weights (edge, node, lrW1, crW1, mrW1) + bias concat
    pack_w<<<640, 256, 0, stream>>>(edgeW, nodeW, lrW1, crW1, mrW1, pack);
    b1cat_k<<<6, 256, 0, stream>>>(lrB1, crB1, mrB1, b1c);

    // 2) gather pair features (averaged) to bf16
    gather_k<<<ROWS, 256, 0, stream>>>(IF, CNF, pairs, xg, ng);

    // 3) feature-resize GEMMs into ci[:, :512] and ci[:, 512:]; mask + bias here
    gemm_wmma<<<dim3(ROWS / 64, 2), 256, 0, stream>>>(
        xg, pack + 0 * REGION_HALVES, edgeB, nrel, ci, KDIM, 0,   0);
    gemm_wmma<<<dim3(ROWS / 64, 2), 256, 0, stream>>>(
        ng, pack + 1 * REGION_HALVES, nodeB, nrel, ci, KDIM, 512, 0);

    // 4) layer-1 for all three heads in one GEMM (N = 1536), ReLU
    gemm_wmma<<<dim3(ROWS / 64, 6), 256, 0, stream>>>(
        ci, pack + 2 * REGION_HALVES, b1c, nullptr, H, 3 * HID, 0, 1);

    // 5) small output layer [512 -> 9|5|3]
    head_out<<<ROWS, 32, 0, stream>>>(H, lrW2, lrB2, crW2, crB2, mrW2, mrB2, out);
}
